// VqaModelDncQC_4346506904358
// MI455X (gfx1250) — matrix operand order repaired and checked by reading, not compiled, not run on hardware
//
#include <hip/hip_runtime.h>
#include <hip/hip_bf16.h>
#include <math.h>

// ---------------- model dims ----------------
#define BATCH   64
#define TSTEPS  20
#define EDIM    300
#define CEDIM   1024
#define ANSDIM  3000
#define HDIM    512
#define NSLOT   256
#define WDIM    64
#define RHEADS  4
#define RWDIM   256            // R*W
#define IFTOT   471
#define CLIPV   20.0f
#define EPSV    1e-6f
#define LDA_INP 1280           // padded row stride for the inp buffer (max of 556 / 1280)

// xi slice offsets (cumsum of IF_SIZES = [256,4,64,1,64,64,4,1,1,12])
#define OFF_RK   0
#define OFF_RS   256
#define OFF_WK   260
#define OFF_WS   324
#define OFF_ER   325
#define OFF_WV   389
#define OFF_FR   453
#define OFF_AG   457
#define OFF_WG   458
#define OFF_MD   459

typedef __attribute__((ext_vector_type(16))) __bf16 v16bf;
typedef __attribute__((ext_vector_type(8)))  float  v8f;

#define ACT_NONE 0
#define ACT_CLIP 1
#define ACT_TANH 2

__device__ __forceinline__ float sigmf_(float x) { return 1.0f / (1.0f + expf(-x)); }
__device__ __forceinline__ float softplusf_(float x) {
  return fmaxf(x, 0.0f) + log1pf(expf(-fabsf(x)));
}

// pack two f32 -> two bf16 in one dword (lowers to v_cvt_pk_bf16_f32)
__device__ __forceinline__ unsigned pack_bf16(float a, float b) {
  union { __bf16 h[2]; unsigned u; } x;
  x.h[0] = (__bf16)a; x.h[1] = (__bf16)b;
  return x.u;
}

// 256-thread block reductions (s must be >=256 floats of shared)
__device__ __forceinline__ float block_reduce_max256(float v, float* s, int tid) {
  s[tid] = v; __syncthreads();
  for (int st = 128; st > 0; st >>= 1) {
    if (tid < st) s[tid] = fmaxf(s[tid], s[tid + st]);
    __syncthreads();
  }
  float r = s[0]; __syncthreads(); return r;
}
__device__ __forceinline__ float block_reduce_sum256(float v, float* s, int tid) {
  s[tid] = v; __syncthreads();
  for (int st = 128; st > 0; st >>= 1) {
    if (tid < st) s[tid] += s[tid + st];
    __syncthreads();
  }
  float r = s[0]; __syncthreads(); return r;
}

// ---------------------------------------------------------------------------
// GEMM: C[64,N] (+=|=) act( A[64,K](f32) @ W[K,N](f32) + bias )
// 256-thread block = 8 waves computes a 64x128 tile, double-buffered LDS.
// Tiles staged coalesced as packed-bf16 pairs, zero-padded at K/N edges
// (branchless). Fragment dwords are CONTIGUOUS in LDS so each fragment is
// two ds_load_b128 straight into WMMA operand quads:
//   A[m][k2] row: pairs for (half) are quads {0-3,8-11} / {4-7,12-15}
//   B stored transposed Bs[n][k2]: fragment = same quads {0-3,8-11} / {4-7,12-15} as A
// All 5 fragments (1 A + 4 B) are loaded into DISTINCT register quads before
// the 4 WMMAs so the compiler can use graduated s_wait_dscnt instead of
// full waits between dependent load/WMMA pairs.
// One barrier per K-step; next tile's global loads issue before the WMMAs.
// Layouts per CDNA5 ISA sec 7.12.2.  M assumed == 64.
// ---------------------------------------------------------------------------
#define LDSTR 20   // row stride in dwords (80B: keeps rows 16B-aligned for B128)

__global__ void __launch_bounds__(256)
gemm_wmma_bf16(const float* __restrict__ A, int lda,
               const float* __restrict__ W, int ldw,
               const float* __restrict__ bias,
               float* __restrict__ C, int ldc,
               int M, int Ncols, int K, int act, int accum)
{
  const int tid  = threadIdx.x;
  const int wave = tid >> 5;
  const int lane = tid & 31;
  const int half = lane >> 4;         // 0: lanes 0-15, 1: lanes 16-31
  const int lidx = lane & 15;

  const int mBand = (wave & 3) * 16;                // 4 M-bands of 16
  const int nHalf = (wave >> 2) * 64;               // 2 N-halves of 64
  const int nTile0 = blockIdx.x * 128;

  __shared__ __align__(16) unsigned As32[2][64][LDSTR];    // [buf][m][k2]
  __shared__ __align__(16) unsigned Bs32[2][128][LDSTR];   // [buf][n][k2] (transposed)

  typedef union { unsigned u[8]; uint4 q[2]; v16bf v; } frag_t;

  v8f acc[4] = {{}, {}, {}, {}};

  // stage one 32-deep K tile into LDS buffer `buf` (clamp+mask, branchless)
  auto stage = [&](int k0, int buf) {
    // A tile: 64 rows x 16 pairs, coalesced
#pragma unroll
    for (int it = 0; it < 4; ++it) {
      int i = tid + it * 256;                 // 0..1023
      int m = i >> 4, p = i & 15;
      int k = k0 + 2 * p;
      int k0c = (k     < K) ? k     : (K - 1);
      int k1c = (k + 1 < K) ? k + 1 : (K - 1);
      float f0 = A[(size_t)m * lda + k0c] * ((k     < K) ? 1.0f : 0.0f);
      float f1 = A[(size_t)m * lda + k1c] * ((k + 1 < K) ? 1.0f : 0.0f);
      As32[buf][m][p] = pack_bf16(f0, f1);
    }
    // B tile: 16 pairs x 128 cols, coalesced along n, stored transposed
#pragma unroll
    for (int it = 0; it < 8; ++it) {
      int i = tid + it * 256;                 // 0..2047
      int p = i >> 7, nn = i & 127;
      int n = nTile0 + nn;
      int k = k0 + 2 * p;
      int nc  = (n     < Ncols) ? n : (Ncols - 1);
      int k0c = (k     < K) ? k     : (K - 1);
      int k1c = (k + 1 < K) ? k + 1 : (K - 1);
      float msk = ((n < Ncols) ? 1.0f : 0.0f);
      float f0 = W[(size_t)k0c * ldw + nc] * (((k     < K) ? 1.0f : 0.0f) * msk);
      float f1 = W[(size_t)k1c * ldw + nc] * (((k + 1 < K) ? 1.0f : 0.0f) * msk);
      Bs32[buf][nn][p] = pack_bf16(f0, f1);
    }
    // prefetch the tile after this one into L2 (global_prefetch_b8)
    int kn = k0 + 32 + lane;
    if (wave == 0 && kn < K) {
      int nc = (nTile0 < Ncols) ? nTile0 : (Ncols - 1);
      __builtin_prefetch(&W[(size_t)kn * ldw + nc], 0, 1);
    }
  };

  const int nsteps = (K + 31) / 32;
  stage(0, 0);

  for (int s = 0; s < nsteps; ++s) {
    const int buf = s & 1;
    __syncthreads();                 // stage(s) visible; compute(s-1) done
    if (s + 1 < nsteps) stage((s + 1) * 32, buf ^ 1);   // overlaps with WMMAs below

    // A fragment: two b128 quads, shared across the 4 N-subtiles
    frag_t af;
    const uint4* arow = (const uint4*)&As32[buf][mBand + lidx][0];
    af.q[0] = arow[half];            // k2 = {0-3} (lo) / {4-7} (hi)
    af.q[1] = arow[2 + half];        // k2 = {8-11} (lo) / {12-15} (hi)

    // all 4 B fragments into distinct quads (graduated dscnt waits)
    frag_t bf[4];
#pragma unroll
    for (int t = 0; t < 4; ++t) {
      int nn = nHalf + t * 16 + lidx;
      const uint4* brow = (const uint4*)&Bs32[buf][nn][0];
      bf[t].q[0] = brow[half];          // k2 = {0-3} (lo) / {4-7} (hi)
      bf[t].q[1] = brow[2 + half];      // k2 = {8-11} (lo) / {12-15} (hi)
    }
#pragma unroll
    for (int t = 0; t < 4; ++t) {
      acc[t] = __builtin_amdgcn_wmma_f32_16x16x32_bf16(
          false, af.v, false, bf[t].v, (short)0, acc[t], false, false);
    }
  }

  // ---- epilogue: bias + activation + (optional) accumulate
#pragma unroll
  for (int t = 0; t < 4; ++t) {
    int n = nTile0 + nHalf + t * 16 + lidx;
    if (n < Ncols) {
      float bv = bias ? bias[n] : 0.0f;
#pragma unroll
      for (int r = 0; r < 8; ++r) {
        int row = mBand + r + 8 * half;     // C/D: VGPR r -> M = r (lo) / 8+r (hi)
        if (row < M) {
          float v = acc[t][r] + bv;
          if (act == ACT_CLIP)      v = fminf(CLIPV, fmaxf(-CLIPV, v));
          else if (act == ACT_TANH) v = tanhf(v);
          float* p = C + (size_t)row * ldc + n;
          *p = accum ? (*p + v) : v;
        }
      }
    }
  }
}

static inline void gemm(hipStream_t s, const float* A, int lda, const float* W, int ldw,
                        const float* bias, float* C, int ldc,
                        int M, int N, int K, int act, int accum)
{
  dim3 g((N + 127) / 128, 1), b(256);
  gemm_wmma_bf16<<<g, b, 0, s>>>(A, lda, W, ldw, bias, C, ldc, M, N, K, act, accum);
}

// ---------------------------------------------------------------------------
// small elementwise / gather kernels
// ---------------------------------------------------------------------------
__global__ void zero_kernel(float* p, unsigned long long n) {
  unsigned long long i = (unsigned long long)blockIdx.x * blockDim.x + threadIdx.x;
  unsigned long long stride = (unsigned long long)gridDim.x * blockDim.x;
  for (; i < n; i += stride) p[i] = 0.0f;
}

// inp[b, 0:300) = tanh(emb[qst[b,t]]); inp[b, 300:556) = rvec[b]
__global__ void build_inp_q(const float* __restrict__ emb, const int* __restrict__ qst,
                            const float* __restrict__ rvec, float* __restrict__ inp, int t)
{
  int idx = blockIdx.x * blockDim.x + threadIdx.x;
  if (idx >= BATCH * 576) return;
  int b = idx / 576, e = idx % 576;
  if (e < EDIM) {
    int tok = qst[b * TSTEPS + t];
    inp[(size_t)b * LDA_INP + e] = tanhf(emb[(size_t)tok * EDIM + e]);
  } else if (e < EDIM + RWDIM) {
    inp[(size_t)b * LDA_INP + e] = rvec[b * RWDIM + (e - EDIM)];
  }
}

__global__ void lstm_kernel(const float* __restrict__ gates, float* __restrict__ h,
                            float* __restrict__ c, float* __restrict__ hc)
{
  int idx = blockIdx.x * blockDim.x + threadIdx.x;
  if (idx >= BATCH * HDIM) return;
  int b = idx / HDIM, j = idx % HDIM;
  const float* g = gates + (size_t)b * 4 * HDIM;
  float gi = g[j], gf = g[HDIM + j], gg = g[2 * HDIM + j], go = g[3 * HDIM + j];
  float cn = sigmf_(gf) * c[idx] + sigmf_(gi) * tanhf(gg);
  float hn = sigmf_(go) * tanhf(cn);
  c[idx] = cn; h[idx] = hn;
  hc[idx] = fminf(CLIPV, fmaxf(-CLIPV, hn));
}

__global__ void build_outin(const float* __restrict__ hc, const float* __restrict__ rvec,
                            float* __restrict__ outin)
{
  int idx = blockIdx.x * blockDim.x + threadIdx.x;
  if (idx >= BATCH * (HDIM + RWDIM)) return;
  int b = idx / (HDIM + RWDIM), j = idx % (HDIM + RWDIM);
  outin[idx] = (j < HDIM) ? hc[b * HDIM + j] : rvec[b * RWDIM + j - HDIM];
}

__global__ void build_last(const float* __restrict__ outc, const float* __restrict__ rvec,
                           float* __restrict__ last)
{
  int idx = blockIdx.x * blockDim.x + threadIdx.x;
  if (idx >= BATCH * (CEDIM + RWDIM)) return;
  int b = idx / (CEDIM + RWDIM), j = idx % (CEDIM + RWDIM);
  float v = (j < CEDIM) ? outc[b * CEDIM + j] : rvec[b * RWDIM + j - CEDIM];
  last[idx] = tanhf(v);
}

// imgf row-L2-normalize, qfeat = qacc/T, inp[b,0:1024)=tanh(imgf_n*qfeat), tail zero
__global__ void combine_kernel(const float* __restrict__ imgf, const float* __restrict__ qacc,
                               float* __restrict__ inp)
{
  int b = blockIdx.x, tid = threadIdx.x;
  __shared__ float sred[256];
  float ss = 0.0f;
  for (int j = tid; j < CEDIM; j += 256) { float v = imgf[b * CEDIM + j]; ss += v * v; }
  sred[tid] = ss; __syncthreads();
  for (int st = 128; st > 0; st >>= 1) { if (tid < st) sred[tid] += sred[tid + st]; __syncthreads(); }
  float nrm = sqrtf(sred[0]);
  for (int j = tid; j < CEDIM; j += 256)
    inp[(size_t)b * LDA_INP + j] =
        tanhf((imgf[b * CEDIM + j] / nrm) * (qacc[b * CEDIM + j] * (1.0f / (float)TSTEPS)));
  for (int j = CEDIM + tid; j < LDA_INP; j += 256) inp[(size_t)b * LDA_INP + j] = 0.0f;
}

// ---------------------------------------------------------------------------
// Fused DNC memory update: one 256-thread workgroup per batch element.
// usage/sort/alloc, write content weights, mem write, link update, prec,
// fw/bw matvecs, read content weights, read weight mix, read vectors.
// ---------------------------------------------------------------------------
__global__ void __launch_bounds__(256)
dnc_mem_kernel(const float* __restrict__ xi,
               float* __restrict__ mem, float* __restrict__ link,
               float* __restrict__ prec, float* __restrict__ wr,
               float* __restrict__ ww, float* __restrict__ usage,
               float* __restrict__ rvec)
{
  const int b = blockIdx.x;
  const int tid = threadIdx.x;

  __shared__ float s_wrold[RHEADS * NSLOT];
  __shared__ float s_cr[RHEADS * NSLOT];
  __shared__ float s_rkeys[RWDIM];
  __shared__ float s_usage[NSLOT];
  __shared__ float s_sorted[NSLOT];
  __shared__ float s_scan[NSLOT];
  __shared__ float s_ww[NSLOT];
  __shared__ float s_prec[NSLOT];
  __shared__ float s_red[NSLOT];
  __shared__ float s_wkey[WDIM], s_erase[WDIM], s_wvec[WDIM];
  __shared__ float s_rstr[RHEADS], s_free[RHEADS], s_knorm[RHEADS];
  __shared__ float s_modes[RHEADS * 3];
  __shared__ float s_sc[4];   // 0: write_str, 1: alloc_g, 2: write_g

  const float* x = xi + (size_t)b * IFTOT;
  float* Mb = mem  + (size_t)b * NSLOT * WDIM;
  float* Lb = link + (size_t)b * NSLOT * NSLOT;

  // ---- parse interface vector ----
  s_rkeys[tid] = tanhf(x[OFF_RK + tid]);
  if (tid < WDIM) {
    s_wkey[tid]  = tanhf(x[OFF_WK + tid]);
    s_erase[tid] = sigmf_(x[OFF_ER + tid]);
    s_wvec[tid]  = tanhf(x[OFF_WV + tid]);
  }
  if (tid < RHEADS) {
    s_rstr[tid] = softplusf_(x[OFF_RS + tid]);
    s_free[tid] = sigmf_(x[OFF_FR + tid]);
    float m0 = x[OFF_MD + 3 * tid], m1 = x[OFF_MD + 3 * tid + 1], m2 = x[OFF_MD + 3 * tid + 2];
    float mx = fmaxf(m0, fmaxf(m1, m2));
    float e0 = expf(m0 - mx), e1 = expf(m1 - mx), e2 = expf(m2 - mx);
    float s = e0 + e1 + e2;
    s_modes[3 * tid]     = e0 / s;
    s_modes[3 * tid + 1] = e1 / s;
    s_modes[3 * tid + 2] = e2 / s;
  }
  if (tid == 0) {
    s_sc[0] = softplusf_(x[OFF_WS]);
    s_sc[1] = sigmf_(x[OFF_AG]);
    s_sc[2] = sigmf_(x[OFF_WG]);
  }
  for (int i = tid; i < RHEADS * NSLOT; i += NSLOT) s_wrold[i] = wr[b * RHEADS * NSLOT + i];
  s_usage[tid] = usage[b * NSLOT + tid];
  s_prec[tid]  = prec[b * NSLOT + tid];
  float ww_old = ww[b * NSLOT + tid];
  __syncthreads();

  if (tid < RHEADS) {
    float kn = 0.0f;
    for (int w = 0; w < WDIM; ++w) { float k = s_rkeys[tid * WDIM + w]; kn += k * k; }
    s_knorm[tid] = sqrtf(kn) + EPSV;
  }

  // ---- usage update: psi, retention ----
  float ps = 1.0f;
  for (int r = 0; r < RHEADS; ++r) ps *= 1.0f - s_free[r] * s_wrold[r * NSLOT + tid];
  float uold = s_usage[tid];
  float u = (uold + ww_old - uold * ww_old) * ps;
  __syncthreads();
  s_usage[tid] = u;
  __syncthreads();

  // ---- rank (stable ascending sort by counting) ----
  int rank = 0;
  for (int j = 0; j < NSLOT; ++j) {
    float uj = s_usage[j];
    rank += (uj < u) || (uj == u && j < tid);
  }
  s_sorted[rank] = u;
  __syncthreads();

  // ---- inclusive product scan (Hillis-Steele) ----
  s_scan[tid] = s_sorted[tid];
  __syncthreads();
  for (int off = 1; off < NSLOT; off <<= 1) {
    float v = s_scan[tid];
    if (tid >= off) v *= s_scan[tid - off];
    __syncthreads();
    s_scan[tid] = v;
    __syncthreads();
  }
  float alloc_n = (1.0f - u) * (rank ? s_scan[rank - 1] : 1.0f);

  // ---- write content weights on OLD memory ----
  float wkn = 0.0f;
  for (int w = 0; w < WDIM; ++w) wkn += s_wkey[w] * s_wkey[w];
  wkn = sqrtf(wkn) + EPSV;
  float dot = 0.0f, mn = 0.0f;
  for (int w = 0; w < WDIM; ++w) {
    float m = Mb[tid * WDIM + w];
    dot += m * s_wkey[w]; mn += m * m;
  }
  float logit = dot / ((sqrtf(mn) + EPSV) * wkn) * s_sc[0];
  float mx = block_reduce_max256(logit, s_red, tid);
  float e = expf(logit - mx);
  float ssum = block_reduce_sum256(e, s_red, tid);
  float cw = e / ssum;

  // ---- write weights ----
  float ag = s_sc[1], wg = s_sc[2];
  float wwn = wg * (ag * alloc_n + (1.0f - ag) * cw);
  s_ww[tid] = wwn;
  ww[b * NSLOT + tid] = wwn;
  float wwsum = block_reduce_sum256(wwn, s_red, tid);   // also barriers s_ww
  float precn = (1.0f - wwsum) * s_prec[tid] + wwn;

  // ---- memory write (own row) ----
  for (int w = 0; w < WDIM; ++w) {
    int idx = tid * WDIM + w;
    Mb[idx] = Mb[idx] * (1.0f - wwn * s_erase[w]) + wwn * s_wvec[w];
  }

  // ---- link matrix update (reads OLD prec from LDS) ----
  for (int idx = tid; idx < NSLOT * NSLOT; idx += NSLOT) {
    int i = idx >> 8, j = idx & (NSLOT - 1);
    float l = (i == j) ? 0.0f
                       : (1.0f - s_ww[i] - s_ww[j]) * Lb[idx] + s_ww[i] * s_prec[j];
    Lb[idx] = l;
  }
  __threadfence_block();   // make mem + link global writes visible block-wide
  __syncthreads();

  // ---- read content weight logits on NEW memory ----
  for (int p = tid; p < RHEADS * NSLOT; p += NSLOT) {
    int r = p >> 8, n = p & (NSLOT - 1);
    const float* mrow = Mb + n * WDIM;
    const float* krow = s_rkeys + r * WDIM;
    float d = 0.0f, m2 = 0.0f;
    for (int w = 0; w < WDIM; ++w) { float m = mrow[w]; d += m * krow[w]; m2 += m * m; }
    s_cr[p] = d / ((sqrtf(m2) + EPSV) * s_knorm[r]) * s_rstr[r];
  }
  __syncthreads();
  for (int r = 0; r < RHEADS; ++r) {
    float lg = s_cr[r * NSLOT + tid];
    float mr = block_reduce_max256(lg, s_red, tid);
    float er = expf(lg - mr);
    float sr = block_reduce_sum256(er, s_red, tid);
    s_cr[r * NSLOT + tid] = er / sr;
    __syncthreads();
  }

  // ---- fw/bw matvecs on NEW link, mix read modes -> new read weights ----
  for (int p = tid; p < RHEADS * NSLOT; p += NSLOT) {
    int r = p >> 8, i = p & (NSLOT - 1);
    const float* wro = s_wrold + r * NSLOT;
    float fwv = 0.0f, bwv = 0.0f;
    for (int j = 0; j < NSLOT; ++j) {
      float wj = wro[j];
      fwv += Lb[i * NSLOT + j] * wj;   // fw = link @ wr_old
      bwv += Lb[j * NSLOT + i] * wj;   // bw = link^T @ wr_old
    }
    float wn = s_modes[3 * r] * bwv + s_modes[3 * r + 1] * s_cr[p] + s_modes[3 * r + 2] * fwv;
    s_cr[p] = wn;                       // reuse s_cr to hold wr_new
    wr[b * RHEADS * NSLOT + p] = wn;
  }
  __syncthreads();

  // ---- read vectors: rvec[r][w] = sum_n wr_new[r][n] * mem[n][w] ----
  {
    int r = tid >> 6, w = tid & 63;
    float rv = 0.0f;
    for (int n = 0; n < NSLOT; ++n) rv += s_cr[r * NSLOT + n] * Mb[n * WDIM + w];
    rvec[b * RWDIM + tid] = rv;
  }
  usage[b * NSLOT + tid] = u;
  prec[b * NSLOT + tid]  = precn;
}

// ---------------------------------------------------------------------------
// host orchestration
// ---------------------------------------------------------------------------
extern "C" void kernel_launch(void* const* d_in, const int* in_sizes, int n_in,
                              void* d_out, int out_size, void* d_ws, size_t ws_size,
                              hipStream_t stream)
{
  (void)in_sizes; (void)n_in; (void)out_size; (void)ws_size;

  const float* img_feat = (const float*)d_in[0];
  const int*   qst      = (const int*)  d_in[1];
  const float* emb      = (const float*)d_in[2];
  const float* img_W    = (const float*)d_in[3];
  const float* img_b    = (const float*)d_in[4];
  const float* q_Wih    = (const float*)d_in[5];
  const float* q_Whh    = (const float*)d_in[6];
  const float* q_b      = (const float*)d_in[7];
  const float* q_Wif    = (const float*)d_in[8];
  const float* q_bif    = (const float*)d_in[9];
  const float* q_Wout   = (const float*)d_in[10];
  const float* q_bout   = (const float*)d_in[11];
  const float* c_Wih    = (const float*)d_in[12];
  const float* c_Whh    = (const float*)d_in[13]; (void)c_Whh; // controller h0==0 -> h@Whh==0
  const float* c_b      = (const float*)d_in[14];
  const float* c_Wif    = (const float*)d_in[15];
  const float* c_bif    = (const float*)d_in[16];
  const float* c_Wout   = (const float*)d_in[17];
  const float* c_bout   = (const float*)d_in[18];
  const float* fc1_W    = (const float*)d_in[19];
  const float* fc1_b    = (const float*)d_in[20];
  const float* fc2_W    = (const float*)d_in[21];
  const float* fc2_b    = (const float*)d_in[22];
  float* out = (float*)d_out;

  // workspace layout (floats); total ~= 6.24M floats (~25 MB)
  float* wsf = (float*)d_ws;
  size_t off = 0;
  auto take = [&](size_t n) { float* p = wsf + off; off += n; return p; };
  float* mem   = take((size_t)BATCH * NSLOT * WDIM);    // 1.0M
  float* link  = take((size_t)BATCH * NSLOT * NSLOT);   // 4.2M
  float* prec  = take((size_t)BATCH * NSLOT);
  float* wr    = take((size_t)BATCH * RHEADS * NSLOT);
  float* wwp   = take((size_t)BATCH * NSLOT);
  float* usg   = take((size_t)BATCH * NSLOT);
  float* rvec  = take((size_t)BATCH * RWDIM);
  float* h     = take((size_t)BATCH * HDIM);
  float* c     = take((size_t)BATCH * HDIM);
  float* qacc  = take((size_t)BATCH * CEDIM);
  const size_t stateN = off;                            // zero-init region
  float* inp   = take((size_t)BATCH * LDA_INP);
  float* gates = take((size_t)BATCH * 4 * HDIM);
  float* hc    = take((size_t)BATCH * HDIM);
  float* xi    = take((size_t)BATCH * IFTOT);
  float* outin = take((size_t)BATCH * (HDIM + RWDIM));
  float* imgf  = take((size_t)BATCH * CEDIM);
  float* outc  = take((size_t)BATCH * CEDIM);
  float* last  = take((size_t)BATCH * (CEDIM + RWDIM));
  float* fc1o  = take((size_t)BATCH * ANSDIM);

  // fresh DNC state + qfeat accumulator every call (deterministic)
  zero_kernel<<<1024, 256, 0, stream>>>(wsf, (unsigned long long)stateN);

  // ---- question encoder: 20 sequential DNC steps ----
  for (int t = 0; t < TSTEPS; ++t) {
    build_inp_q<<<(BATCH * 576 + 255) / 256, 256, 0, stream>>>(emb, qst, rvec, inp, t);
    gemm(stream, inp, LDA_INP, q_Wih, 4 * HDIM, q_b, gates, 4 * HDIM,
         BATCH, 4 * HDIM, EDIM + RWDIM, ACT_NONE, 0);
    gemm(stream, h, HDIM, q_Whh, 4 * HDIM, nullptr, gates, 4 * HDIM,
         BATCH, 4 * HDIM, HDIM, ACT_NONE, 1);
    lstm_kernel<<<(BATCH * HDIM + 255) / 256, 256, 0, stream>>>(gates, h, c, hc);
    gemm(stream, hc, HDIM, q_Wif, IFTOT, q_bif, xi, IFTOT,
         BATCH, IFTOT, HDIM, ACT_NONE, 0);
    dnc_mem_kernel<<<BATCH, NSLOT, 0, stream>>>(xi, mem, link, prec, wr, wwp, usg, rvec);
    build_outin<<<(BATCH * (HDIM + RWDIM) + 255) / 256, 256, 0, stream>>>(hc, rvec, outin);
    // qfeat accumulator: qacc += clip(outin @ Wout + bout)
    gemm(stream, outin, HDIM + RWDIM, q_Wout, CEDIM, q_bout, qacc, CEDIM,
         BATCH, CEDIM, HDIM + RWDIM, ACT_CLIP, 1);
  }

  // ---- image path + combine (qfeat = qacc / T inside combine) ----
  gemm(stream, img_feat, 4096, img_W, CEDIM, img_b, imgf, CEDIM,
       BATCH, CEDIM, 4096, ACT_NONE, 0);
  combine_kernel<<<BATCH, 256, 0, stream>>>(imgf, qacc, inp);

  // ---- controller DNC: fresh state, single step on comb ----
  zero_kernel<<<1024, 256, 0, stream>>>(wsf, (unsigned long long)stateN);
  gemm(stream, inp, LDA_INP, c_Wih, 4 * HDIM, c_b, gates, 4 * HDIM,
       BATCH, 4 * HDIM, CEDIM + RWDIM, ACT_NONE, 0);     // h==0 -> Whh term omitted
  lstm_kernel<<<(BATCH * HDIM + 255) / 256, 256, 0, stream>>>(gates, h, c, hc);
  gemm(stream, hc, HDIM, c_Wif, IFTOT, c_bif, xi, IFTOT,
       BATCH, IFTOT, HDIM, ACT_NONE, 0);
  dnc_mem_kernel<<<BATCH, NSLOT, 0, stream>>>(xi, mem, link, prec, wr, wwp, usg, rvec);
  build_outin<<<(BATCH * (HDIM + RWDIM) + 255) / 256, 256, 0, stream>>>(hc, rvec, outin);
  gemm(stream, outin, HDIM + RWDIM, c_Wout, CEDIM, c_bout, outc, CEDIM,
       BATCH, CEDIM, HDIM + RWDIM, ACT_CLIP, 0);

  // ---- head: last = tanh(concat(out_c, rv_c)); fc1 tanh; fc2 ----
  build_last<<<(BATCH * (CEDIM + RWDIM) + 255) / 256, 256, 0, stream>>>(outc, rvec, last);
  gemm(stream, last, CEDIM + RWDIM, fc1_W, ANSDIM, fc1_b, fc1o, ANSDIM,
       BATCH, ANSDIM, CEDIM + RWDIM, ACT_TANH, 0);
  gemm(stream, fc1o, ANSDIM, fc2_W, ANSDIM, fc2_b, out, ANSDIM,
       BATCH, ANSDIM, ANSDIM, ACT_NONE, 0);
}
